// LDStack_74234214744597
// MI455X (gfx1250) — matrix operand order repaired and checked by reading, not compiled, not run on hardware
//
#include <hip/hip_runtime.h>
#include <hip/hip_bf16.h>

// ---------------------------------------------------------------------------
// LDStack for MI455X (gfx1250, wave32, WMMA) — time-segmented version.
//
// The output einsum is a (16384 x 2048)x(2048 x 128) bf16-WMMA GEMM (8.6 GFLOP);
// HBM traffic ~16 MB -> the real constraint is occupancy, since the scan is
// time-sequential. The recurrence is linear, so it is decomposed over S=16
// segments of L=64 steps: local ends + carry factors per segment, cheap O(S)
// chains for the inbound states, then an exact per-segment replay fused with
// the WMMA GEMM. Heavy kernels run 256 workgroups instead of 16.
// bf16 WMMA w/ f32 accum is safe: the s*Cp term is ~1e-4 of the exact-f32 xc*D term.
// ---------------------------------------------------------------------------

typedef __attribute__((ext_vector_type(16))) __bf16 v16bf;
typedef __attribute__((ext_vector_type(8)))  __bf16 v8bf;
typedef __attribute__((ext_vector_type(8)))  float  v8f;

#define T_LEN   1024
#define B_SZ    16
#define D_DIM   128
#define K_PROJ  16
#define N_EIG   64
#define M_OUT   128
#define CHN     1024          // K_PROJ * N_EIG channels
#define KPAD    2056          // LDS A row stride (bf16): 4112 B -> 16B bank step/row
#define CHUNK   16
#define SEGS    16            // time segments
#define SEGL    64            // steps per segment
#define SCHUNK  (SEGL / CHUNK)

// workspace layout (bytes)
#define WS_LAM   0                     // float[2*CHN]            8 KB
#define WS_B     8192                  // float[2*CHN]            8 KB
#define WS_W     16384                 // __bf16[2048*128]      512 KB
#define WS_XC    540672                // float[B*T*K_PROJ]       1 MB
#define WS_E1    1589248               // float[2*B*SEGS*CHN]     2 MB
#define WS_S1IN  3686400               // float[2*B*SEGS*CHN]     2 MB
#define WS_E2    5783552               // float[2*B*SEGS*CHN]     2 MB
#define WS_P2    7880704               // float[2*B*SEGS*CHN]     2 MB
#define WS_S2IN  9977856               // float[2*B*SEGS*CHN]     2 MB (end ~12 MB)

#define SEG_IDX(b, s, c) (((size_t)(b) * SEGS + (s)) * CHN + (c))

// dynamic LDS layout for ld_main (bytes)
#define SM_XCC   0                     // float[16*17]        = 1088
#define SM_DBUF  1088                  // float[16*128]       = 8192
#define SM_DOV   9280                  // float[128]          = 512
#define SM_RED   9792                  // float[3*8*256]      = 24576
#define SM_ABUF  34368                 // __bf16[16*KPAD]     = 65792
#define SMEM_BYTES 100160

// ---------------------------------------------------------------------------
// K1: lambda = exp(+/- i theta); B[c] = 1 / prod_{i!=j} (1 - lam_i/lam_j)
// ---------------------------------------------------------------------------
__global__ void ld_lamB(const float* __restrict__ theta,
                        float* __restrict__ lam,
                        float* __restrict__ Bc) {
  int c = blockIdx.x * blockDim.x + threadIdx.x;
  if (c >= CHN) return;
  int k = c >> 6, j = c & 63;
  const float* th = theta + k * 32;
  float tj = th[j & 31];
  float sj = (j < 32) ? 1.f : -1.f;
  float lx = cosf(tj), ly = sj * sinf(tj);
  lam[2 * c] = lx;
  lam[2 * c + 1] = ly;

  float px = 1.f, py = 0.f;
  for (int i = 0; i < N_EIG; ++i) {
    if (i == j) continue;
    float ti = th[i & 31];
    float si = (i < 32) ? 1.f : -1.f;
    float lix = cosf(ti), liy = si * sinf(ti);
    float rx = lix * lx + liy * ly;      // lam_i * conj(lam_j)
    float ry = liy * lx - lix * ly;
    float tx = 1.f - rx, ty = -ry;
    float nx = px * tx - py * ty;
    float ny = px * ty + py * tx;
    px = nx; py = ny;
  }
  float d2 = px * px + py * py;
  Bc[2 * c] = px / d2;
  Bc[2 * c + 1] = -py / d2;
}

// ---------------------------------------------------------------------------
// K2: Cp[k,j,m] = sum_i C[k,m,i]*lam_j^{-(63-i)}; pack bf16 B-operand blocks:
// block (mt,kb) 32(K)x16(N), element (kk,mm) at ((mt*64+kb)*16+mm)*32+kk.
// 1/16 (mean over k) folded in; -Im folded so a plain dot gives Re(y).
// ---------------------------------------------------------------------------
__global__ void ld_weights(const float* __restrict__ C,
                           const float* __restrict__ lam,
                           __bf16* __restrict__ Wt) {
  int gid = blockIdx.x * blockDim.x + threadIdx.x;
  if (gid >= CHN * M_OUT) return;
  int c = gid >> 7, m = gid & 127;
  int k = c >> 6;
  float cjx = lam[2 * c];
  float cjy = -lam[2 * c + 1];
  const float* Crow = C + ((size_t)k * M_OUT + m) * N_EIG;
  float px = 1.f, py = 0.f;
  float ar = 0.f, ai = 0.f;
  for (int i = N_EIG - 1; i >= 0; --i) {
    float cv = Crow[i];
    ar = fmaf(cv, px, ar);
    ai = fmaf(cv, py, ai);
    float nx = px * cjx - py * cjy;
    float ny = px * cjy + py * cjx;
    px = nx; py = ny;
  }
  int kb = c >> 5, kk = c & 31, mt = m >> 4, mm = m & 15;
  Wt[((size_t)(mt * 64 + kb) * 16 + mm) * 32 + kk] = (__bf16)(ar * 0.0625f);
  Wt[((size_t)(mt * 64 + (kb + 32)) * 16 + mm) * 32 + kk] = (__bf16)(-ai * 0.0625f);
}

// ---------------------------------------------------------------------------
// K3 (phase 1): per (b,seg): project xc rows [t0-1, t0+63] into LDS (row -1
// is zero), cache canonical rows [t0, t0+63] in ws, and compute the s1
// segment-local end state E1 (zero inbound state).
// ---------------------------------------------------------------------------
__global__ __launch_bounds__(1024, 1)
void ld_phase1(const float* __restrict__ x,
               const float* __restrict__ Rm,
               const float* __restrict__ lam,
               const float* __restrict__ Bcoef,
               float* __restrict__ xcws,
               float* __restrict__ E1) {
  __shared__ float xcl[(SEGL + 1) * 17];
  const int tid = threadIdx.x;
  const int seg = blockIdx.x, b = blockIdx.y;
  const int t0 = seg * SEGL;

  for (int i = tid; i < (SEGL + 1) * K_PROJ; i += 1024) {
    int r = i >> 4, kk = i & 15;
    int t = t0 - 1 + r;
    float acc = 0.f;
    if (t >= 0) {
      const float* xrow = x + ((size_t)b * T_LEN + t) * D_DIM;
      #pragma unroll 4
      for (int d = 0; d < D_DIM; ++d) acc = fmaf(xrow[d], Rm[d * K_PROJ + kk], acc);
      if (r >= 1) xcws[((size_t)b * T_LEN + t) * K_PROJ + kk] = acc;
    }
    xcl[r * 17 + kk] = acc;
  }
  __syncthreads();

  const int kq = tid >> 6;
  const float lamx = lam[2 * tid], lamy = lam[2 * tid + 1];
  const float bcx = Bcoef[2 * tid], bcy = Bcoef[2 * tid + 1];
  float sx = 0.f, sy = 0.f;
  for (int tt = 0; tt < SEGL; ++tt) {           // xcl row tt == xc[t-1]
    float xcv = xcl[tt * 17 + kq];
    float bxx = xcv * bcx, bxy = xcv * bcy;
    float nx = lamx * sx - lamy * sy + bxx;
    float ny = lamx * sy + lamy * sx + bxy;
    sx = nx; sy = ny;
  }
  size_t o = 2 * SEG_IDX(b, seg, tid);
  E1[o] = sx; E1[o + 1] = sy;
}

// ---------------------------------------------------------------------------
// K4 (phase 2): chain s1 inbound states: S1in[0]=0,
// S1in[s] = lam^L * S1in[s-1] + E1[s-1].  lam^64 by 6 squarings.
// ---------------------------------------------------------------------------
__global__ __launch_bounds__(1024, 1)
void ld_chain1(const float* __restrict__ lam,
               const float* __restrict__ E1,
               float* __restrict__ S1in) {
  const int c = threadIdx.x, b = blockIdx.x;
  float lx = lam[2 * c], ly = lam[2 * c + 1];
  #pragma unroll
  for (int q = 0; q < 6; ++q) {                 // lam^(2^6) = lam^64
    float nx = lx * lx - ly * ly;
    float ny = 2.f * lx * ly;
    lx = nx; ly = ny;
  }
  float cx = 0.f, cy = 0.f;
  for (int s = 0; s < SEGS; ++s) {
    size_t o = 2 * SEG_IDX(b, s, c);
    S1in[o] = cx; S1in[o + 1] = cy;
    float ex = E1[o], ey = E1[o + 1];
    float nx = lx * cx - ly * cy + ex;
    float ny = lx * cy + ly * cx + ey;
    cx = nx; cy = ny;
  }
}

// ---------------------------------------------------------------------------
// K5 (phase 3): per (b,seg): replay true s1 from S1in -> alpha known ->
// s2 segment-local end E2 (zero inbound) and carry product P2 = prod(alpha*lam).
// ---------------------------------------------------------------------------
__global__ __launch_bounds__(1024, 1)
void ld_phase3(const float* __restrict__ xcws,
               const float* __restrict__ lam,
               const float* __restrict__ Bcoef,
               const float* __restrict__ S1in,
               float* __restrict__ E2,
               float* __restrict__ P2) {
  __shared__ float xcl[(SEGL + 1) * 17];
  const int tid = threadIdx.x;
  const int seg = blockIdx.x, b = blockIdx.y;
  const int t0 = seg * SEGL;

  for (int i = tid; i < (SEGL + 1) * K_PROJ; i += 1024) {
    int r = i >> 4, kk = i & 15;
    int t = t0 - 1 + r;
    xcl[r * 17 + kk] = (t >= 0) ? xcws[((size_t)b * T_LEN + t) * K_PROJ + kk] : 0.f;
  }
  __syncthreads();

  const int kq = tid >> 6;
  const float lamx = lam[2 * tid], lamy = lam[2 * tid + 1];
  const float bcx = Bcoef[2 * tid], bcy = Bcoef[2 * tid + 1];
  size_t o = 2 * SEG_IDX(b, seg, tid);
  float s1x = S1in[o], s1y = S1in[o + 1];
  float s2x = 0.f, s2y = 0.f;
  float Px = 1.f, Py = 0.f;
  for (int tt = 0; tt < SEGL; ++tt) {
    float xcv = xcl[tt * 17 + kq];
    float bxx = xcv * bcx, bxy = xcv * bcy;     // Bx[t-1]
    float alpha = rsqrtf(fmaf(s1x, s1x, fmaf(s1y, s1y, 1.f)));
    float ar = alpha * lamx, ai = alpha * lamy; // a2[t]
    float nx = ar * s2x - ai * s2y + bxx;
    float ny = ar * s2y + ai * s2x + bxy;
    s2x = nx; s2y = ny;
    nx = ar * Px - ai * Py;
    ny = ar * Py + ai * Px;
    Px = nx; Py = ny;
    nx = lamx * s1x - lamy * s1y + bxx;
    ny = lamx * s1y + lamy * s1x + bxy;
    s1x = nx; s1y = ny;
  }
  E2[o] = s2x; E2[o + 1] = s2y;
  P2[o] = Px;  P2[o + 1] = Py;
}

// ---------------------------------------------------------------------------
// K6 (phase 4): chain s2 inbound states: S2in[s] = P2[s-1]*S2in[s-1] + E2[s-1].
// ---------------------------------------------------------------------------
__global__ __launch_bounds__(1024, 1)
void ld_chain2(const float* __restrict__ E2,
               const float* __restrict__ P2,
               float* __restrict__ S2in) {
  const int c = threadIdx.x, b = blockIdx.x;
  float cx = 0.f, cy = 0.f;
  for (int s = 0; s < SEGS; ++s) {
    size_t o = 2 * SEG_IDX(b, s, c);
    S2in[o] = cx; S2in[o + 1] = cy;
    float px = P2[o], py = P2[o + 1];
    float nx = px * cx - py * cy + E2[o];
    float ny = px * cy + py * cx + E2[o + 1];
    cx = nx; cy = ny;
  }
}

// ---------------------------------------------------------------------------
// K7 (phase 5): per (b,seg): exact replay from (S1in,S2in), 4 chunks of 16
// steps; bf16 s2 staged in LDS; 32 waves run v_wmma_f32_16x16x32_bf16 with
// wave (mt=w>>2, ks=w&3); slice-0 reduces K-partials, adds xc*D/16+Do, stores.
// ---------------------------------------------------------------------------
__global__ __launch_bounds__(1024, 1)
void ld_main(const float* __restrict__ xcws,
             const float* __restrict__ D,
             const float* __restrict__ Do,
             const float* __restrict__ lam,
             const float* __restrict__ Bcoef,
             const float* __restrict__ S1in,
             const float* __restrict__ S2in,
             const __bf16* __restrict__ Wt,
             float* __restrict__ out) {
  extern __shared__ char smem[];
  float*  xcc  = (float*)(smem + SM_XCC);     // [16][17]
  float*  Dbuf = (float*)(smem + SM_DBUF);    // [16][128], pre-scaled 1/16
  float*  Dov  = (float*)(smem + SM_DOV);     // [128]
  float*  red  = (float*)(smem + SM_RED);     // [3][8][256]
  __bf16* Abuf = (__bf16*)(smem + SM_ABUF);   // [16][KPAD]

  const int tid  = threadIdx.x;
  const int seg  = blockIdx.x, b = blockIdx.y;
  const int t0   = seg * SEGL;
  const int lane = tid & 31;
  const int wave = tid >> 5;
  const int mt   = wave >> 2;
  const int ks   = wave & 3;
  const int kq   = tid >> 6;

  for (int i = tid; i < K_PROJ * M_OUT; i += 1024) Dbuf[i] = D[i] * 0.0625f;
  if (tid < M_OUT) Dov[tid] = Do[tid];

  const float lamx = lam[2 * tid], lamy = lam[2 * tid + 1];
  const float bcx  = Bcoef[2 * tid], bcy = Bcoef[2 * tid + 1];
  size_t o = 2 * SEG_IDX(b, seg, tid);
  float s1x = S1in[o], s1y = S1in[o + 1];
  float s2x = S2in[o], s2y = S2in[o + 1];
  float xm1 = (t0 > 0) ? xcws[((size_t)b * T_LEN + t0 - 1) * K_PROJ + kq] : 0.f;
  float bxx = xm1 * bcx, bxy = xm1 * bcy;       // Bx[t0-1]

  for (int cc = 0; cc < SCHUNK; ++cc) {
    const int tc = t0 + cc * CHUNK;
    __syncthreads();
    if (tid < 256) {
      int tt = tid >> 4, kk = tid & 15;
      xcc[tt * 17 + kk] = xcws[((size_t)b * T_LEN + tc + tt) * K_PROJ + kk];
    }
    __syncthreads();

    #pragma unroll 4
    for (int tt = 0; tt < CHUNK; ++tt) {        // fused two-iteration scan
      float xcv = xcc[tt * 17 + kq];
      float alpha = rsqrtf(fmaf(s1x, s1x, fmaf(s1y, s1y, 1.f)));
      float lr = alpha * lamx, li = alpha * lamy;
      float nx = lr * s2x - li * s2y + bxx;
      float ny = lr * s2y + li * s2x + bxy;
      s2x = nx; s2y = ny;
      nx = lamx * s1x - lamy * s1y + bxx;
      ny = lamx * s1y + lamy * s1x + bxy;
      s1x = nx; s1y = ny;
      bxx = xcv * bcx; bxy = xcv * bcy;
      Abuf[tt * KPAD + tid]       = (__bf16)s2x;
      Abuf[tt * KPAD + CHN + tid] = (__bf16)s2y;
    }
    __syncthreads();

    v8f acc = {};
    const int row = lane & 15;
    const int hsel = lane >> 4;
    #pragma unroll 2
    for (int kb = ks * 16; kb < ks * 16 + 16; ++kb) {
      int klo = kb * 32 + hsel * 16;
      v8bf alo = *(const v8bf*)&Abuf[row * KPAD + klo];
      v8bf ahi = *(const v8bf*)&Abuf[row * KPAD + klo + 8];
      v16bf a = __builtin_shufflevector(alo, ahi,
          0, 1, 2, 3, 4, 5, 6, 7, 8, 9, 10, 11, 12, 13, 14, 15);
      const v16bf bm = *(const v16bf*)(Wt +
          ((size_t)(mt * 64 + kb) * 16 + row) * 32 + hsel * 16);
      acc = __builtin_amdgcn_wmma_f32_16x16x32_bf16(
          false, a, false, bm, (short)0, acc, false, false);
    }
    if (ks != 0) {
      float* rp = red + ((ks - 1) * 8 + mt) * 256;
      #pragma unroll
      for (int r = 0; r < 8; ++r) rp[r * 32 + lane] = acc[r];
    }
    __syncthreads();

    if (ks == 0) {
      #pragma unroll
      for (int j = 0; j < 3; ++j) {
        const float* rp = red + (j * 8 + mt) * 256;
        #pragma unroll
        for (int r = 0; r < 8; ++r) acc[r] += rp[r * 32 + lane];
      }
      const int mm = lane & 15;
      const int m  = mt * 16 + mm;
      #pragma unroll
      for (int r = 0; r < 8; ++r) {
        int tr = r + hsel * 8;
        float dterm = 0.f;
        #pragma unroll
        for (int kk2 = 0; kk2 < K_PROJ; ++kk2)
          dterm = fmaf(xcc[tr * 17 + kk2], Dbuf[kk2 * M_OUT + m], dterm);
        out[((size_t)b * T_LEN + tc + tr) * M_OUT + m] = acc[r] + dterm + Dov[m];
      }
    }
  }
}

extern "C" void kernel_launch(void* const* d_in, const int* in_sizes, int n_in,
                              void* d_out, int out_size, void* d_ws, size_t ws_size,
                              hipStream_t stream) {
  (void)in_sizes; (void)n_in; (void)out_size; (void)ws_size;
  const float* x     = (const float*)d_in[0];
  const float* Rm    = (const float*)d_in[1];
  const float* theta = (const float*)d_in[2];
  const float* C     = (const float*)d_in[3];
  const float* D     = (const float*)d_in[4];
  const float* Do    = (const float*)d_in[5];
  float* out = (float*)d_out;

  char* ws = (char*)d_ws;
  float*  lamA = (float*)(ws + WS_LAM);
  float*  BcA  = (float*)(ws + WS_B);
  __bf16* Wt   = (__bf16*)(ws + WS_W);
  float*  xcws = (float*)(ws + WS_XC);
  float*  E1   = (float*)(ws + WS_E1);
  float*  S1in = (float*)(ws + WS_S1IN);
  float*  E2   = (float*)(ws + WS_E2);
  float*  P2   = (float*)(ws + WS_P2);
  float*  S2in = (float*)(ws + WS_S2IN);

  dim3 segGrid(SEGS, B_SZ);
  ld_lamB<<<dim3(CHN / 256), dim3(256), 0, stream>>>(theta, lamA, BcA);
  ld_weights<<<dim3((CHN * M_OUT) / 256), dim3(256), 0, stream>>>(C, lamA, Wt);
  ld_phase1<<<segGrid, dim3(1024), 0, stream>>>(x, Rm, lamA, BcA, xcws, E1);
  ld_chain1<<<dim3(B_SZ), dim3(1024), 0, stream>>>(lamA, E1, S1in);
  ld_phase3<<<segGrid, dim3(1024), 0, stream>>>(xcws, lamA, BcA, S1in, E2, P2);
  ld_chain2<<<dim3(B_SZ), dim3(1024), 0, stream>>>(E2, P2, S2in);
  ld_main<<<segGrid, dim3(1024), SMEM_BYTES, stream>>>(
      xcws, D, Do, lamA, BcA, S1in, S2in, Wt, out);
}
